// SelfAttention_10024453669243
// MI455X (gfx1250) — matrix operand order repaired and checked by reading, not compiled, not run on hardware
//
#include <hip/hip_runtime.h>
#include <hip/hip_bf16.h>

// ---------------------------------------------------------------------------
// Causal self-attention for B=2, T=2048, C=1024, H=16, D=64 on gfx1250.
// Stage 1: QKV projection GEMM (f32 in -> f16 Q/K/V in ws), WMMA f16,
//          double-buffered LDS + global prefetch.
// Stage 2: flash-attention per (b,h,64-row q tile), WMMA f16, Q/K tiles
//          staged with the Tensor Data Mover when available, f32 out.
// ---------------------------------------------------------------------------

typedef __attribute__((ext_vector_type(16))) _Float16     v16h;
typedef __attribute__((ext_vector_type(8)))  float        v8f;
typedef __attribute__((ext_vector_type(4)))  unsigned int v4u;
typedef __attribute__((ext_vector_type(4)))  int          v4i;
typedef __attribute__((ext_vector_type(8)))  int          v8i;

union Frag16 {          // one WMMA 16-bit A/B operand (16 halves = 8 VGPRs)
    v16h  v;
    uint4 u[2];
};

__device__ __forceinline__ v8f wmma_f16(const Frag16& a, const Frag16& b, v8f c) {
    return __builtin_amdgcn_wmma_f32_16x16x32_f16(
        /*neg_a=*/false, a.v, /*neg_b=*/false, b.v,
        /*c_mod=*/(short)0, c, /*reuse_a=*/false, /*reuse_b=*/false);
}

#define SA_B    2
#define SA_T    2048
#define SA_C    1024
#define SA_H    16
#define SA_D    64
#define SA_M    (SA_B * SA_T)     // 4096 rows of x
#define SA_N3   (3 * SA_C)        // 3072 qkv columns
#define SA_K    SA_C              // 1024 reduction

// ---- Tensor Data Mover availability (clang-23 / therock 6-arg form) -------
#if defined(__has_builtin)
# if __has_builtin(__builtin_amdgcn_tensor_load_to_lds) && \
     __has_builtin(__builtin_amdgcn_s_wait_tensorcnt)
#  define SA_HAVE_TDM 1
# endif
#endif
#ifndef SA_HAVE_TDM
# define SA_HAVE_TDM 0
#endif

#if SA_HAVE_TDM
// Issue one TDM 2D tile load: 64x64 f16, contiguous rows of 64 elements in
// global memory, padded in LDS to a 72-half row pitch (pad 4 dwords every
// 32 dwords). Wave-level op (EXEC ignored); call from one wave only.
__device__ __forceinline__ void sa_tdm_load_64x64_f16(const _Float16* gsrc,
                                                      void* lds_dst) {
    const unsigned long long ga = (unsigned long long)(uintptr_t)gsrc;
    const unsigned int lds     = (unsigned int)(uintptr_t)lds_dst; // LDS byte offset

    v4u g0;
    g0.x = 1u;                                   // count=1, user descriptor
    g0.y = lds;                                  // lds_addr
    g0.z = (unsigned int)ga;                     // global_addr[31:0]
    g0.w = (unsigned int)(ga >> 32) | 0x80000000u; // global_addr[56:32] | type=2

    v8i g1;
    // data_size=2B (code 1), pad_enable, pad_interval=32 dwords (code 4),
    // pad_amount=4 dwords (code 3), no multicast / atomic-barrier / iterate.
    g1[0] = (int)((1u << 16) | (1u << 20) | (4u << 22) | (3u << 25));
    g1[1] = (int)(64u << 16);                    // tensor_dim0 = 64 (lo16)
    g1[2] = (int)(64u << 16);                    // dim0 hi=0 | tensor_dim1 = 64
    g1[3] = (int)(64u << 16);                    // dim1 hi=0 | tile_dim0 = 64
    g1[4] = 64;                                  // tile_dim1 = 64, tile_dim2 = 0
    g1[5] = 64;                                  // tensor_dim0_stride = 64 elems
    g1[6] = 0;
    g1[7] = 0;

    v4i gz4 = {0, 0, 0, 0};                      // groups 2/3 unused (2D tensor)
    v8i gz8 = {0, 0, 0, 0, 0, 0, 0, 0};          // extra group (clang-23 form)
    __builtin_amdgcn_tensor_load_to_lds(g0, g1, gz4, gz4, gz8, 0);
}
#endif

// ===========================================================================
// Kernel 1: qkv = x @ w_qkv  (M=4096, N=3072, K=1024), f16 output into ws
// laid out as qkv[s][b][h][t][d], s in {q,k,v}. Q pre-scaled by 1/sqrt(D).
// Block: 256 threads (8 waves). Tile 128x128, k-step 32, double-buffered LDS.
// Wave (wm in 0..3, wn in 0..1) computes rows wm*32..+31, cols wn*64..+63.
// ===========================================================================
__global__ __launch_bounds__(256)
void sa_qkv_gemm(const float* __restrict__ x,
                 const float* __restrict__ w,
                 _Float16* __restrict__ qkv) {
    __shared__ _Float16 As[2][128][40];   // [m][k], pitch 40 halves (16B-aligned rows)
    __shared__ _Float16 Bs[2][128][40];   // [n][k] (transposed), pitch 40

    const int tid  = threadIdx.x;
    const int lane = tid & 31;
    const int wv   = tid >> 5;     // 0..7
    const int wm   = wv >> 1;      // 0..3
    const int wn   = wv & 1;       // 0..1
    const int ln   = lane & 15;
    const int half = lane >> 4;
    const int bm   = blockIdx.y * 128;
    const int bn   = blockIdx.x * 128;

    // Per-thread staging assignment (fixed across k-steps).
    const int arow = tid >> 1, aseg = tid & 1;   // A: 16 cols of one row
    const int bkr  = tid >> 3, bseg = tid & 7;   // B: 16 cols of one k-row
    const float* aptr = x + (size_t)(bm + arow) * SA_K + aseg * 16;
    const float* bptr = w + (size_t)bkr * SA_N3 + bn + bseg * 16;

    float ra[16], rb[16];
#pragma unroll
    for (int i = 0; i < 16; ++i) ra[i] = aptr[i];
#pragma unroll
    for (int i = 0; i < 16; ++i) rb[i] = bptr[i];

    // store tile 0 into buffer 0
    {
        union { _Float16 h[16]; uint4 q[2]; } pk;
#pragma unroll
        for (int i = 0; i < 16; ++i) pk.h[i] = (_Float16)ra[i];
        *(uint4*)&As[0][arow][aseg * 16]     = pk.q[0];
        *(uint4*)&As[0][arow][aseg * 16 + 8] = pk.q[1];
#pragma unroll
        for (int i = 0; i < 16; ++i) Bs[0][bseg * 16 + i][bkr] = (_Float16)rb[i];
    }
    __syncthreads();

    v8f acc[2][4];
#pragma unroll
    for (int mt = 0; mt < 2; ++mt)
#pragma unroll
        for (int nt = 0; nt < 4; ++nt)
#pragma unroll
            for (int r = 0; r < 8; ++r) acc[mt][nt][r] = 0.0f;

    int buf = 0;
    for (int k0 = 0; k0 < SA_K; k0 += 32, buf ^= 1) {
        const bool nxt = (k0 + 32) < SA_K;
        if (nxt) {
            const float* an = aptr + (k0 + 32);
            const float* bp = bptr + (size_t)(k0 + 32) * SA_N3;
            __builtin_prefetch(an + 32, 0, 0);                 // 2 k-steps ahead
            __builtin_prefetch(bp + (size_t)32 * SA_N3, 0, 0);
#pragma unroll
            for (int i = 0; i < 16; ++i) ra[i] = an[i];
#pragma unroll
            for (int i = 0; i < 16; ++i) rb[i] = bp[i];
        }

        // ---- fragments from current buffer + 8 WMMAs ----
        Frag16 a[2], b[4];
#pragma unroll
        for (int mt = 0; mt < 2; ++mt) {
            const int r = wm * 32 + mt * 16 + ln;              // A: lane = row M
            a[mt].u[0] = *(const uint4*)&As[buf][r][half * 8];       // K = half*8..+7
            a[mt].u[1] = *(const uint4*)&As[buf][r][16 + half * 8];  // K = 16+half*8..+7
        }
#pragma unroll
        for (int nt = 0; nt < 4; ++nt) {
            const int n = wn * 64 + nt * 16 + ln;              // B: lane = col N
            b[nt].u[0] = *(const uint4*)&Bs[buf][n][half * 8];       // K = half*8..+7
            b[nt].u[1] = *(const uint4*)&Bs[buf][n][16 + half * 8];  // K = 16+half*8..+7
        }
#pragma unroll
        for (int mt = 0; mt < 2; ++mt)
#pragma unroll
            for (int nt = 0; nt < 4; ++nt)
                acc[mt][nt] = wmma_f16(a[mt], b[nt], acc[mt][nt]);

        // ---- stage next tile into the other buffer ----
        if (nxt) {
            union { _Float16 h[16]; uint4 q[2]; } pk;
#pragma unroll
            for (int i = 0; i < 16; ++i) pk.h[i] = (_Float16)ra[i];
            *(uint4*)&As[buf ^ 1][arow][aseg * 16]     = pk.q[0];
            *(uint4*)&As[buf ^ 1][arow][aseg * 16 + 8] = pk.q[1];
#pragma unroll
            for (int i = 0; i < 16; ++i)
                Bs[buf ^ 1][bseg * 16 + i][bkr] = (_Float16)rb[i];
            __syncthreads();
        }
    }

    // ---- epilogue: scatter f16 into qkv[s][b][h][t][d]; scale Q by 0.125 ----
    const int   sec = bn >> 10;                     // 0=Q,1=K,2=V (128 | 1024)
    const float qs  = (sec == 0) ? 0.125f : 1.0f;   // 1/sqrt(64)
#pragma unroll
    for (int mt = 0; mt < 2; ++mt) {
        const int rg0 = bm + wm * 32 + mt * 16 + half * 8;    // C: M = r + 8*half
#pragma unroll
        for (int nt = 0; nt < 4; ++nt) {
            const int cg  = bn + wn * 64 + nt * 16 + ln;      // C: N = lane&15
            const int rem = cg & 1023;
            const int hh  = rem >> 6;
            const int dd  = rem & 63;
#pragma unroll
            for (int r = 0; r < 8; ++r) {
                const int rg = rg0 + r;
                const int bb = rg >> 11;       // batch (M rows = b*2048+t)
                const int tt = rg & 2047;
                qkv[(((size_t)sec * SA_B + bb) * SA_H + hh) * ((size_t)SA_T * SA_D)
                    + (size_t)tt * SA_D + dd] = (_Float16)(acc[mt][nt][r] * qs);
            }
        }
    }
}

// ===========================================================================
// Kernel 2: flash attention. Grid (qt = T/64, bh = B*H). Block 128 (4 waves).
// Wave w owns query rows w*16..+15 of the 64-row tile; online softmax.
// Q/K tiles staged by the Tensor Data Mover when available.
// ===========================================================================
__global__ __launch_bounds__(128)
void sa_attn(const _Float16* __restrict__ qkv, float* __restrict__ out) {
    __shared__ _Float16 Qs[64][72];       // [q][d]
    __shared__ _Float16 Ks[64][72];       // [key][d]
    __shared__ _Float16 Vt[64][72];       // [d][key]   (transposed V)
    __shared__ _Float16 Ps[4][16][72];    // per-wave P 16x64 (C -> A relayout)

    const int tid  = threadIdx.x;
    const int lane = tid & 31;
    const int w    = tid >> 5;            // 0..3
    const int ln   = lane & 15;
    const int half = lane >> 4;
    const int qt   = blockIdx.x;          // query tile (64 rows)
    const int bh   = blockIdx.y;
    const int b    = bh >> 4;
    const int h    = bh & 15;

    const size_t plane = (size_t)SA_T * SA_D;
    const _Float16* Q = qkv + (((size_t)0 * SA_B + b) * SA_H + h) * plane + (size_t)qt * 64 * SA_D;
    const _Float16* K = qkv + (((size_t)1 * SA_B + b) * SA_H + h) * plane;
    const _Float16* V = qkv + (((size_t)2 * SA_B + b) * SA_H + h) * plane;

    // ---- load Q tile once ----
#if SA_HAVE_TDM
    if (tid < 32) {                            // wave 0 issues the DMA
        sa_tdm_load_64x64_f16(Q, &Qs[0][0]);
    }
    __builtin_amdgcn_s_wait_tensorcnt(0);
#else
    {
        const int row = tid >> 1, off = (tid & 1) * 32;
        const uint4* src = (const uint4*)(Q + (size_t)row * SA_D + off);
        uint4 q0 = src[0], q1 = src[1], q2 = src[2], q3 = src[3];
        uint4* dst = (uint4*)&Qs[row][off];
        dst[0] = q0; dst[1] = q1; dst[2] = q2; dst[3] = q3;
    }
#endif
    __syncthreads();

    // ---- Q A-fragments, held for whole key loop (2 k-steps over d) ----
    Frag16 qf[2];
#pragma unroll
    for (int kd = 0; kd < 2; ++kd) {
        const int r = w * 16 + ln;
        qf[kd].u[0] = *(const uint4*)&Qs[r][kd * 32 + half * 8];
        qf[kd].u[1] = *(const uint4*)&Qs[r][kd * 32 + 16 + half * 8];
    }

    float m_i[8], l_i[8];
    v8f o[4];
#pragma unroll
    for (int r = 0; r < 8; ++r) { m_i[r] = -1e30f; l_i[r] = 0.0f; }
#pragma unroll
    for (int j = 0; j < 4; ++j)
#pragma unroll
        for (int r = 0; r < 8; ++r) o[j][r] = 0.0f;

    for (int kt = 0; kt <= qt; ++kt) {
        __syncthreads();   // previous K/V tile fully consumed before overwrite

        // ---- stage K tile (TDM if available) and V tile transposed ----
#if SA_HAVE_TDM
        if (tid < 32) {
            sa_tdm_load_64x64_f16(K + (size_t)kt * 64 * SA_D, &Ks[0][0]);
        }
#else
        {
            const int row = tid >> 1, off = (tid & 1) * 32;
            const uint4* ks = (const uint4*)(K + (size_t)(kt * 64 + row) * SA_D + off);
            uint4 k0 = ks[0], k1 = ks[1], k2 = ks[2], k3 = ks[3];
            uint4* kd_ = (uint4*)&Ks[row][off];
            kd_[0] = k0; kd_[1] = k1; kd_[2] = k2; kd_[3] = k3;
        }
#endif
        {
            const int row = tid >> 1, off = (tid & 1) * 32;
            const uint4* vs = (const uint4*)(V + (size_t)(kt * 64 + row) * SA_D + off);
            union { _Float16 hh[32]; uint4 q[4]; } pv;
            pv.q[0] = vs[0]; pv.q[1] = vs[1]; pv.q[2] = vs[2]; pv.q[3] = vs[3];
#pragma unroll
            for (int i = 0; i < 32; ++i) Vt[off + i][row] = pv.hh[i];
        }
#if SA_HAVE_TDM
        __builtin_amdgcn_s_wait_tensorcnt(0);
#endif
        __syncthreads();

        // ---- S = Q * K^T  (M=16 rows/wave, N=64 keys, K=64 d) ----
        v8f s[4];
#pragma unroll
        for (int j = 0; j < 4; ++j)
#pragma unroll
            for (int r = 0; r < 8; ++r) s[j][r] = 0.0f;
#pragma unroll
        for (int kd = 0; kd < 2; ++kd) {
#pragma unroll
            for (int j = 0; j < 4; ++j) {
                Frag16 kf;
                const int n = j * 16 + ln;                     // key index = B col
                kf.u[0] = *(const uint4*)&Ks[n][kd * 32 + half * 8];
                kf.u[1] = *(const uint4*)&Ks[n][kd * 32 + 16 + half * 8];
                s[j] = wmma_f16(qf[kd], kf, s[j]);
            }
        }

        // ---- causal mask on diagonal tile ----
        if (kt == qt) {
#pragma unroll
            for (int j = 0; j < 4; ++j)
#pragma unroll
                for (int r = 0; r < 8; ++r)
                    if (j * 16 + ln > w * 16 + half * 8 + r) s[j][r] = -1e30f;
        }

        // ---- online softmax (rows live in one 16-lane half) ----
#pragma unroll
        for (int r = 0; r < 8; ++r) {
            float mx = fmaxf(fmaxf(s[0][r], s[1][r]), fmaxf(s[2][r], s[3][r]));
#pragma unroll
            for (int off = 1; off < 16; off <<= 1)
                mx = fmaxf(mx, __shfl_xor(mx, off, 32));
            const float mn  = fmaxf(m_i[r], mx);
            const float scl = __expf(m_i[r] - mn);
            m_i[r] = mn;
            float rs = 0.0f;
#pragma unroll
            for (int j = 0; j < 4; ++j) {
                const float p = __expf(s[j][r] - mn);
                s[j][r] = p;
                rs += p;
            }
#pragma unroll
            for (int off = 1; off < 16; off <<= 1)
                rs += __shfl_xor(rs, off, 32);
            l_i[r] = l_i[r] * scl + rs;
#pragma unroll
            for (int j = 0; j < 4; ++j) o[j][r] *= scl;
        }

        // ---- P: C-layout -> LDS -> A-layout (wave-private region) ----
#pragma unroll
        for (int j = 0; j < 4; ++j)
#pragma unroll
            for (int r = 0; r < 8; ++r)
                Ps[w][half * 8 + r][j * 16 + ln] = (_Float16)s[j][r];

        // ---- O += P * V  (K=64 keys, N=64 d) ----
#pragma unroll
        for (int kd2 = 0; kd2 < 2; ++kd2) {
            Frag16 pf;
            pf.u[0] = *(const uint4*)&Ps[w][ln][kd2 * 32 + half * 8];
            pf.u[1] = *(const uint4*)&Ps[w][ln][kd2 * 32 + 16 + half * 8];
#pragma unroll
            for (int j = 0; j < 4; ++j) {
                Frag16 vf;
                const int n = j * 16 + ln;                     // d index = B col
                vf.u[0] = *(const uint4*)&Vt[n][kd2 * 32 + half * 8];
                vf.u[1] = *(const uint4*)&Vt[n][kd2 * 32 + 16 + half * 8];
                o[j] = wmma_f16(pf, vf, o[j]);
            }
        }
    }

    // ---- normalize + store f32 out[b][t][h*64+d] ----
#pragma unroll
    for (int r = 0; r < 8; ++r) {
        const float inv = 1.0f / l_i[r];
        const int   t   = qt * 64 + w * 16 + half * 8 + r;
#pragma unroll
        for (int j = 0; j < 4; ++j) {
            const int c = h * SA_D + j * 16 + ln;
            out[((size_t)b * SA_T + t) * SA_C + c] = o[j][r] * inv;
        }
    }
}

// ===========================================================================
extern "C" void kernel_launch(void* const* d_in, const int* in_sizes, int n_in,
                              void* d_out, int out_size, void* d_ws, size_t ws_size,
                              hipStream_t stream) {
    (void)in_sizes; (void)n_in; (void)out_size;
    const float* x = (const float*)d_in[0];      // [2,2048,1024] f32
    const float* w = (const float*)d_in[1];      // [1024,3072] f32
    float* out     = (float*)d_out;              // [2,2048,1024] f32
    _Float16* qkv  = (_Float16*)d_ws;            // [3][B][H][T][D] f16

    const size_t need = (size_t)3 * SA_B * SA_H * SA_T * SA_D * sizeof(_Float16);
    if (ws_size < need) return;                  // 24 MiB of scratch required

    dim3 g1(SA_N3 / 128, SA_M / 128);            // (24, 32)
    sa_qkv_gemm<<<g1, 256, 0, stream>>>(x, w, qkv);

    dim3 g2(SA_T / 64, SA_B * SA_H);             // (32, 32)
    sa_attn<<<g2, 128, 0, stream>>>(qkv, out);
}
